// ThornsAttention_38242388804057
// MI455X (gfx1250) — compile-verified, matrix operand order checked by reading, not run on hardware
//
#include <hip/hip_runtime.h>
#include <hip/hip_bf16.h>

typedef __attribute__((ext_vector_type(16))) __bf16 v16bf;
typedef __attribute__((ext_vector_type(8)))  float  v8f;
typedef unsigned int v4u __attribute__((ext_vector_type(4)));
typedef int          v8i __attribute__((ext_vector_type(8)));
typedef int          v4i __attribute__((ext_vector_type(4)));

#define BATCH 2
#define SEQ   2048
#define DIM   1024
#define NHEAD 16
#define HDIM  64

// ---------------------------------------------------------------------------
// Fragment loader for 16-bit A/B operands of v_wmma_*_16x16x32_bf16.
// Per ISA 7.12.2: lane L holds row (L&15); element e maps to
// K = (e&7) + 8*( ((e>>3)<<1) | (L>>4) ).
// => two ds_load_b128 per fragment.
// ---------------------------------------------------------------------------
static __device__ __forceinline__ v16bf frag_ld(const __bf16* rowbase, int half) {
    union { v16bf v; uint4 u[2]; } r;
    r.u[0] = *reinterpret_cast<const uint4*>(rowbase + half * 8);
    r.u[1] = *reinterpret_cast<const uint4*>(rowbase + 16 + half * 8);
    return r.v;
}

// ---------------------------------------------------------------------------
// Tensor Data Mover: DMA one 16-row x 64-element bf16 tile (row pitch
// `stride_elems`) from global memory into LDS at byte offset `lds_off`,
// padding +8 elements after each 64-element row (LDS row stride 72 elems),
// matching frag_ld's expected layout.  D# per ISA 08_async_tensor §8.3-8.5:
//   g0: count=1 | lds_addr | global_addr[56:0] | type=2
//   g1: data_size=2B, pad_enable, pad_interval=32dw, pad_amount=4dw,
//       tensor_dim0=64 @bits79:48, tensor_dim1=16 @bits111:80,
//       tile_dim0=64 @127:112, tile_dim1=16 @143:128,
//       tensor_dim0_stride @207:160
//   g2/g3: zero (2-D tensor); trailing int32x8 operand (clang-23 6-arg form)
//   is unused state for 2-D tiles -> zeros; cpol=0.
// Per-wave issue; completion via s_wait_tensorcnt.
// ---------------------------------------------------------------------------
static __device__ __forceinline__ void tdm_load_tile16x64(unsigned int lds_off,
                                                          const __bf16* gsrc,
                                                          unsigned int stride_elems) {
    unsigned long long ga = (unsigned long long)(uintptr_t)gsrc;
    v4u g0;
    g0[0] = 1u;                                              // count=1, user D#
    g0[1] = lds_off;                                         // LDS byte address
    g0[2] = (unsigned int)ga;                                // global addr [31:0]
    g0[3] = (unsigned int)((ga >> 32) & 0x1FFFFFFull) | (2u << 30);  // [56:32] | type=2
    v8i g1;
    g1[0] = (int)((1u << 16) | (1u << 20) | (4u << 22) | (3u << 25));
    //             data=2B     pad_en       ival=32dw     amount=4dw
    g1[1] = (int)(64u << 16);   // tensor_dim0 = 64 (bits 79:48, low half)
    g1[2] = (int)(16u << 16);   // tensor_dim0 hi=0 | tensor_dim1 = 16 (low half)
    g1[3] = (int)(64u << 16);   // tensor_dim1 hi=0 | tile_dim0 = 64
    g1[4] = 16;                 // tile_dim1 = 16 | tile_dim2 = 0
    g1[5] = (int)stride_elems;  // tensor_dim0_stride [31:0]
    g1[6] = 0;                  // stride hi | tensor_dim1_stride lo
    g1[7] = 0;
    v4i g2 = {0, 0, 0, 0};
    v4i g3 = {0, 0, 0, 0};
    v8i g4 = {0, 0, 0, 0, 0, 0, 0, 0};
    __builtin_amdgcn_tensor_load_to_lds(g0, g1, g2, g3, g4, 0);
}

// ---------------------------------------------------------------------------
// fp32 -> bf16 cast, 4 elements/thread (exact grids only)
// ---------------------------------------------------------------------------
__global__ __launch_bounds__(256) void cast_f32_bf16(const float* __restrict__ in,
                                                     __bf16* __restrict__ out, int n) {
    int i = (blockIdx.x * blockDim.x + threadIdx.x) * 4;
    if (i + 3 < n) {
        float4 v = *reinterpret_cast<const float4*>(in + i);
        union { __bf16 e[4]; uint2 u; } s;
        s.e[0] = (__bf16)v.x; s.e[1] = (__bf16)v.y;
        s.e[2] = (__bf16)v.z; s.e[3] = (__bf16)v.w;
        *reinterpret_cast<uint2*>(out + i) = s.u;
    }
}

// ---------------------------------------------------------------------------
// C[M,N] = A[M,K] @ W[N,K]^T   (bf16 in, f32 accumulate)
// 128x128 tile, K-step 64, double-buffered LDS filled by TDM (wave w DMAs its
// 16-row slice of both tiles).  16 WMMAs per wave per barrier.
// ---------------------------------------------------------------------------
#define GTM 128
#define GTN 128
#define GTK 64
#define GSTR 72   // 64 + 8 pad elements (produced by TDM pad feature)

template <bool OUT_F32>
__global__ __launch_bounds__(256) void gemm_bf16(const __bf16* __restrict__ A,
                                                 const __bf16* __restrict__ W,
                                                 void* __restrict__ Out,
                                                 int M, int N, int K) {
    __shared__ __align__(16) __bf16 smem[2 * GTM * GSTR + 2 * GTN * GSTR];

    const int tid  = threadIdx.x;
    const int lane = tid & 31;
    const int wid  = tid >> 5;
    const int half = lane >> 4;
    const int l16  = lane & 15;
    const int m0   = blockIdx.x * GTM;
    const int n0   = blockIdx.y * GTN;
    const int wm   = (wid & 3) * 32;
    const int wn   = (wid >> 2) * 64;

    const unsigned int ABYTES = (unsigned int)(GTM * GSTR * 2);
    const unsigned int WBASE  = 2u * ABYTES;
    const unsigned int WBYTES = (unsigned int)(GTN * GSTR * 2);
    const unsigned int SLICE  = (unsigned int)(wid * 16 * GSTR * 2);

    v8f acc[2][4];
#pragma unroll
    for (int i = 0; i < 2; ++i)
#pragma unroll
        for (int j = 0; j < 4; ++j)
#pragma unroll
            for (int r = 0; r < 8; ++r) acc[i][j][r] = 0.f;

    // prologue: DMA stage 0 into buffer 0
    tdm_load_tile16x64(SLICE,         A + (size_t)(m0 + wid * 16) * K, K);
    tdm_load_tile16x64(WBASE + SLICE, W + (size_t)(n0 + wid * 16) * K, K);
    __builtin_amdgcn_s_wait_tensorcnt(0);
    __syncthreads();

    const int NST = K / GTK;
    for (int ks = 0; ks < NST; ++ks) {
        const int buf = ks & 1;
        if (ks + 1 < NST) {
            const int k0 = (ks + 1) * GTK;
            tdm_load_tile16x64((buf ^ 1) * ABYTES + SLICE,
                               A + (size_t)(m0 + wid * 16) * K + k0, K);
            tdm_load_tile16x64(WBASE + (buf ^ 1) * WBYTES + SLICE,
                               W + (size_t)(n0 + wid * 16) * K + k0, K);
        }

        const __bf16* As = smem + buf * (GTM * GSTR);
        const __bf16* Ws = smem + 2 * (GTM * GSTR) + buf * (GTN * GSTR);
        v16bf af[2][2], bfr[4][2];
#pragma unroll
        for (int i = 0; i < 2; ++i)
#pragma unroll
            for (int kc = 0; kc < 2; ++kc)
                af[i][kc] = frag_ld(As + (wm + i * 16 + l16) * GSTR + kc * 32, half);
#pragma unroll
        for (int j = 0; j < 4; ++j)
#pragma unroll
            for (int kc = 0; kc < 2; ++kc)
                bfr[j][kc] = frag_ld(Ws + (wn + j * 16 + l16) * GSTR + kc * 32, half);

#pragma unroll
        for (int kc = 0; kc < 2; ++kc)
#pragma unroll
            for (int i = 0; i < 2; ++i)
#pragma unroll
                for (int j = 0; j < 4; ++j)
                    acc[i][j] = __builtin_amdgcn_wmma_f32_16x16x32_bf16(
                        false, af[i][kc], false, bfr[j][kc], (short)0, acc[i][j], false, false);

        if (ks + 1 < NST) __builtin_amdgcn_s_wait_tensorcnt(0);
        __syncthreads();
    }

    // D layout: reg r -> row r + 8*(lane>>4), col lane&15
#pragma unroll
    for (int i = 0; i < 2; ++i)
#pragma unroll
        for (int j = 0; j < 4; ++j)
#pragma unroll
            for (int r = 0; r < 8; ++r) {
                size_t row = (size_t)(m0 + wm + i * 16 + r + 8 * half);
                int col = n0 + wn + j * 16 + l16;
                if constexpr (OUT_F32)
                    reinterpret_cast<float*>(Out)[row * N + col] = acc[i][j][r];
                else
                    reinterpret_cast<__bf16*>(Out)[row * N + col] = (__bf16)acc[i][j][r];
            }
}

// ---------------------------------------------------------------------------
// Fused flash attention with ALiBi + causal mask.  Q,K,V,O bf16 [B,S,D].
// grid = (S/128, NHEAD, BATCH), 256 threads = 8 waves, wave w owns q-rows
// [qb*128 + w*16, +16).  K blocks are DMA'd by the TDM (double-buffered, one
// 16x64 slice per wave); V blocks use VGPR staging + scalar LDS transpose.
// One barrier per k-block.  Q staging LDS is overlaid with the P buffer.
// ---------------------------------------------------------------------------
__global__ __launch_bounds__(256) void attn_kernel(const __bf16* __restrict__ Q,
                                                   const __bf16* __restrict__ Kg,
                                                   const __bf16* __restrict__ V,
                                                   __bf16* __restrict__ O) {
    const int qb = blockIdx.x;
    const int h  = blockIdx.y;
    const int b  = blockIdx.z;
    const int tid  = threadIdx.x;
    const int lane = tid & 31;
    const int w    = tid >> 5;
    const int half = lane >> 4;
    const int l16  = lane & 15;
    const float slope = exp2f(-0.5f * (float)(h + 1));  // 1/((2^8)^((h+1)/16))

    // LDS: P (overlaid with Q staging) + double-buffered K + double-buffered V^T
    __shared__ __align__(16) __bf16 smem[128 * 136 + 2 * 128 * 72 + 2 * 64 * 136];
    __bf16* Ps  = smem;                       // 128 x 136 (rows = q, cols = keys)
    __bf16* Qs  = smem;                       // overlay: 128 x 72, pre-loop only
    __bf16* KsB = smem + 128 * 136;           // 2 x (128 x 72)
    __bf16* VtB = smem + 128 * 136 + 2 * 128 * 72;  // 2 x (64 x 136) transposed
    const unsigned int LDS_KS = 128u * 136u * 2u;   // byte offset of KsB
    const unsigned int KSZ    = 128u * 72u * 2u;    // bytes per K buffer

    const size_t headoff = (size_t)b * SEQ * DIM + (size_t)h * HDIM;
    const int lrow = tid >> 1;
    const int lseg = (tid & 1) * 32;  // 32 bf16 = 64B per thread

    // ---- load Q block, hoist Q fragments ----
    {
        const uint4* g = reinterpret_cast<const uint4*>(Q + headoff + (size_t)(qb * 128 + lrow) * DIM + lseg);
        uint4* s = reinterpret_cast<uint4*>(Qs + lrow * 72 + lseg);
        s[0] = g[0]; s[1] = g[1]; s[2] = g[2]; s[3] = g[3];
    }
    __syncthreads();
    v16bf qf[2];
    qf[0] = frag_ld(Qs + (w * 16 + l16) * 72 + 0,  half);
    qf[1] = frag_ld(Qs + (w * 16 + l16) * 72 + 32, half);

    // ---- prologue: K block 0 via TDM, V block 0 via regs+transpose ----
    tdm_load_tile16x64(LDS_KS + (unsigned int)(w * 16 * 72 * 2),
                       Kg + headoff + (size_t)(w * 16) * DIM, DIM);
    uint4 rv[4];
    {
        const uint4* gv = reinterpret_cast<const uint4*>(V + headoff + (size_t)lrow * DIM + lseg);
#pragma unroll
        for (int u = 0; u < 4; ++u) rv[u] = gv[u];
#pragma unroll
        for (int q4 = 0; q4 < 4; ++q4) {
            union { uint4 u; __bf16 e[8]; } tv; tv.u = rv[q4];
            int c0 = lseg + q4 * 8;
#pragma unroll
            for (int e = 0; e < 8; ++e) VtB[(c0 + e) * 136 + lrow] = tv.e[e];
        }
    }

    float mrow[8], lrowv[8];
    v8f oacc[4];
#pragma unroll
    for (int r = 0; r < 8; ++r) { mrow[r] = -1e30f; lrowv[r] = 0.f; }
#pragma unroll
    for (int t = 0; t < 4; ++t)
#pragma unroll
        for (int r = 0; r < 8; ++r) oacc[t][r] = 0.f;

    __builtin_amdgcn_s_wait_tensorcnt(0);
    __syncthreads();  // Q frags read; buffer 0 complete; Ps region now free

    for (int kb = 0; kb <= qb; ++kb) {
        const int buf = kb & 1;
        const __bf16* Ks = KsB + buf * (128 * 72);
        const __bf16* Vt = VtB + buf * (64 * 136);

        // async prefetch of next K block (TDM) + next V block (regs)
        if (kb < qb) {
            tdm_load_tile16x64(LDS_KS + (unsigned int)(buf ^ 1) * KSZ + (unsigned int)(w * 16 * 72 * 2),
                               Kg + headoff + (size_t)((kb + 1) * 128 + w * 16) * DIM, DIM);
            const uint4* gv = reinterpret_cast<const uint4*>(V + headoff + (size_t)((kb + 1) * 128 + lrow) * DIM + lseg);
#pragma unroll
            for (int u = 0; u < 4; ++u) rv[u] = gv[u];
        }

        // ---- scores = Q @ K^T : 16 rows x 128 keys per wave ----
        v8f sfr[8];
#pragma unroll
        for (int nt = 0; nt < 8; ++nt)
#pragma unroll
            for (int r = 0; r < 8; ++r) sfr[nt][r] = 0.f;
#pragma unroll
        for (int kc = 0; kc < 2; ++kc) {
            v16bf kf[8];
#pragma unroll
            for (int nt = 0; nt < 8; ++nt)
                kf[nt] = frag_ld(Ks + (nt * 16 + l16) * 72 + kc * 32, half);
#pragma unroll
            for (int nt = 0; nt < 8; ++nt)
                sfr[nt] = __builtin_amdgcn_wmma_f32_16x16x32_bf16(
                              false, qf[kc], false, kf[nt], (short)0, sfr[nt], false, false);
        }

        // ---- scale + ALiBi + causal mask, block row max ----
        float bmax[8];
#pragma unroll
        for (int r = 0; r < 8; ++r) bmax[r] = -1e30f;
        const int qbase = qb * 128 + w * 16 + 8 * half;
        const int kcol0 = kb * 128 + l16;
#pragma unroll
        for (int nt = 0; nt < 8; ++nt) {
            int kj = kcol0 + nt * 16;
#pragma unroll
            for (int r = 0; r < 8; ++r) {
                int qi = qbase + r;
                float sv = sfr[nt][r] * 0.125f - slope * (float)(kj - qi);
                if (kj > qi) sv = -1e30f;
                sfr[nt][r] = sv;
                bmax[r] = fmaxf(bmax[r], sv);
            }
        }
#pragma unroll
        for (int r = 0; r < 8; ++r) {
            float v = bmax[r];
#pragma unroll
            for (int off = 1; off < 16; off <<= 1) v = fmaxf(v, __shfl_xor(v, off, 32));
            bmax[r] = v;
        }

        // ---- online softmax update ----
        float rsum[8];
#pragma unroll
        for (int r = 0; r < 8; ++r) {
            float mnew = fmaxf(mrow[r], bmax[r]);
            float sc = __expf(mrow[r] - mnew);
            mrow[r] = mnew;
            lrowv[r] *= sc;
#pragma unroll
            for (int t = 0; t < 4; ++t) oacc[t][r] *= sc;
            rsum[r] = 0.f;
        }
#pragma unroll
        for (int nt = 0; nt < 8; ++nt)
#pragma unroll
            for (int r = 0; r < 8; ++r) {
                float p = __expf(sfr[nt][r] - mrow[r]);
                sfr[nt][r] = p;
                rsum[r] += p;
            }
#pragma unroll
        for (int r = 0; r < 8; ++r) {
            float v = rsum[r];
#pragma unroll
            for (int off = 1; off < 16; off <<= 1) v += __shfl_xor(v, off, 32);
            lrowv[r] += v;
        }

        // ---- P -> bf16 into this wave's private LDS rows (DS in-order/wave) ----
#pragma unroll
        for (int nt = 0; nt < 8; ++nt)
#pragma unroll
            for (int r = 0; r < 8; ++r)
                Ps[(w * 16 + r + 8 * half) * 136 + nt * 16 + l16] = (__bf16)sfr[nt][r];

        // ---- O += P @ V : 4 d-tiles, K = 128 keys in 4 chunks ----
#pragma unroll
        for (int kc = 0; kc < 4; ++kc) {
            v16bf pa = frag_ld(Ps + (w * 16 + l16) * 136 + kc * 32, half);
            v16bf vb[4];
#pragma unroll
            for (int t = 0; t < 4; ++t)
                vb[t] = frag_ld(Vt + (t * 16 + l16) * 136 + kc * 32, half);
#pragma unroll
            for (int t = 0; t < 4; ++t)
                oacc[t] = __builtin_amdgcn_wmma_f32_16x16x32_bf16(
                              false, pa, false, vb[t], (short)0, oacc[t], false, false);
        }

        // ---- stage next V block; ensure next K DMA has landed ----
        if (kb < qb) {
            __bf16* Vtn = VtB + (buf ^ 1) * (64 * 136);
#pragma unroll
            for (int q4 = 0; q4 < 4; ++q4) {
                union { uint4 u; __bf16 e[8]; } tv; tv.u = rv[q4];
                int c0 = lseg + q4 * 8;
#pragma unroll
                for (int e = 0; e < 8; ++e) Vtn[(c0 + e) * 136 + lrow] = tv.e[e];
            }
            __builtin_amdgcn_s_wait_tensorcnt(0);
        }
        __syncthreads();
    }

    // ---- epilogue: normalize, write bf16 [B,S,D] ----
#pragma unroll
    for (int t = 0; t < 4; ++t)
#pragma unroll
        for (int r = 0; r < 8; ++r) {
            int qi = qb * 128 + w * 16 + r + 8 * half;
            float o = oacc[t][r] / lrowv[r];
            O[headoff + (size_t)qi * DIM + t * 16 + l16] = (__bf16)o;
        }
}

// ---------------------------------------------------------------------------
extern "C" void kernel_launch(void* const* d_in, const int* in_sizes, int n_in,
                              void* d_out, int out_size, void* d_ws, size_t ws_size,
                              hipStream_t stream) {
    const float* x  = (const float*)d_in[0];
    const float* Wq = (const float*)d_in[1];
    const float* Wk = (const float*)d_in[2];
    const float* Wv = (const float*)d_in[3];
    const float* Wo = (const float*)d_in[4];

    const size_t MK = (size_t)BATCH * SEQ * DIM;  // 4,194,304
    const size_t WW = (size_t)DIM * DIM;          // 1,048,576

    __bf16* p  = (__bf16*)d_ws;
    __bf16* xb = p; p += MK;
    __bf16* wq = p; p += WW;
    __bf16* wk = p; p += WW;
    __bf16* wv = p; p += WW;
    __bf16* wo = p; p += WW;
    __bf16* Qb = p; p += MK;
    __bf16* Kb = p; p += MK;
    __bf16* Vb = p; p += MK;
    __bf16* Ab = p; p += MK;   // attention output, [B,S,D] bf16

    cast_f32_bf16<<<dim3(MK / 1024), dim3(256), 0, stream>>>(x,  xb, (int)MK);
    cast_f32_bf16<<<dim3(WW / 1024), dim3(256), 0, stream>>>(Wq, wq, (int)WW);
    cast_f32_bf16<<<dim3(WW / 1024), dim3(256), 0, stream>>>(Wk, wk, (int)WW);
    cast_f32_bf16<<<dim3(WW / 1024), dim3(256), 0, stream>>>(Wv, wv, (int)WW);
    cast_f32_bf16<<<dim3(WW / 1024), dim3(256), 0, stream>>>(Wo, wo, (int)WW);

    dim3 gg(32, 8), gb(256);
    const int M = BATCH * SEQ, N = DIM, K = DIM;
    gemm_bf16<false><<<gg, gb, 0, stream>>>(xb, wq, Qb, M, N, K);
    gemm_bf16<false><<<gg, gb, 0, stream>>>(xb, wk, Kb, M, N, K);
    gemm_bf16<false><<<gg, gb, 0, stream>>>(xb, wv, Vb, M, N, K);

    attn_kernel<<<dim3(SEQ / 128, NHEAD, BATCH), dim3(256), 0, stream>>>(Qb, Kb, Vb, Ab);

    gemm_bf16<true><<<gg, gb, 0, stream>>>(Ab, wo, d_out, M, N, K);
}
